// NeuralNet_32658931319021
// MI455X (gfx1250) — matrix operand order repaired and checked by reading, not compiled, not run on hardware
//
#include <hip/hip_runtime.h>
#include <math.h>

// ---------------- problem dims ----------------
static constexpr int Mm   = 100;
static constexpr int Cc   = 30;
static constexpr int Dd   = 300;
static constexpr int Kk   = 3;
static constexpr int DP   = 320;   // padded 300 (reduction dim, mult of 32)
static constexpr int NEP  = 928;   // padded 900 (reduction dim for f GEMM)
static constexpr int ROWP = 112;   // padded 100 (mult of 16)
static constexpr int COLP = 304;   // padded 300 (mult of 16)
static constexpr int MC   = 3000;  // M*C
static constexpr int MCP  = 3008;  // padded (mult of 16)
static constexpr int T_MC  = MCP / 16;   // 188
static constexpr int T_COL = COLP / 16;  // 19
static constexpr int T_ROW = ROWP / 16;  // 7

typedef __attribute__((ext_vector_type(16))) __bf16 v16bf;
typedef __attribute__((ext_vector_type(8)))  __bf16 v8bf;
typedef __attribute__((ext_vector_type(8)))  float  v8f;

// ---------------- bf16 helpers ----------------
__device__ __forceinline__ unsigned short f2bf(float x) {
  unsigned int u = __float_as_uint(x);
  u += 0x7FFFu + ((u >> 16) & 1u);   // round-to-nearest-even
  return (unsigned short)(u >> 16);
}

// A-fragment (16x32 bf16, contiguous k order, same as B):
// lane<16: M=lane,   K = kc+0..15
// lane>=16:M=lane-16,K = kc+16..31
__device__ __forceinline__ v16bf load_a_frag(const unsigned short* A, int lda,
                                             int rowBase, int kc, int lane) {
  int m  = rowBase + (lane & 15);
  int kb = kc + ((lane & 16) ? 16 : 0);
  const unsigned short* p = A + (size_t)m * lda + kb;
  v8bf lo = *reinterpret_cast<const v8bf*>(p);
  v8bf hi = *reinterpret_cast<const v8bf*>(p + 8);
  v16bf r;
#pragma unroll
  for (int i = 0; i < 8; ++i) { r[i] = lo[i]; r[8 + i] = hi[i]; }
  return r;
}

// B-fragment (32x16 bf16): lane<16: N=lane, K=kc+0..15 ; lane>=16: N=lane-16, K=kc+16..31
// B stored row-major as [N][K] (NT GEMM: C = A * B^T), so this is one contiguous 32B load.
__device__ __forceinline__ v16bf load_b_frag(const unsigned short* B, int ldb,
                                             int colBase, int kc, int lane) {
  int n  = colBase + (lane & 15);
  int kb = kc + ((lane & 16) ? 16 : 0);
  return *reinterpret_cast<const v16bf*>(B + (size_t)n * ldb + kb);
}

__device__ __forceinline__ v8f wmma_acc(v16bf a, v16bf b, v8f c) {
  return __builtin_amdgcn_wmma_f32_16x16x32_bf16(false, a, false, b, (short)0, c,
                                                 false, false);
}

__device__ __forceinline__ v8f wmma_nt_tile(const unsigned short* A, int lda,
                                            const unsigned short* B, int ldb,
                                            int rowBase, int colBase, int kTot,
                                            int lane) {
  v8f acc = {0.f, 0.f, 0.f, 0.f, 0.f, 0.f, 0.f, 0.f};
  for (int kc = 0; kc < kTot; kc += 32) {
    v16bf af = load_a_frag(A, lda, rowBase, kc, lane);
    v16bf bf = load_b_frag(B, ldb, colBase, kc, lane);
    acc = wmma_acc(af, bf, acc);
  }
  return acc;
}

// ---------------- utility kernels ----------------
__global__ void zero_u32_kernel(unsigned int* p, int n) {
  int t = blockIdx.x * blockDim.x + threadIdx.x;
  if (t < n) p[t] = 0u;
}

__global__ void conv_ent_kernel(const float* ent, unsigned short* entbf) {
  int t = blockIdx.x * blockDim.x + threadIdx.x;           // mc*300 + d
  if (t >= MC * Dd) return;
  int mc = t / Dd, d = t % Dd;
  entbf[(size_t)mc * DP + d] = f2bf(ent[t]);
}

__global__ void conv_fmc_kernel(const float* x, unsigned short* xbf) {
  int t = blockIdx.x * blockDim.x + threadIdx.x;           // m*900 + d
  if (t >= Mm * 900) return;
  int m = t / 900, d = t % 900;
  xbf[(size_t)m * NEP + d] = f2bf(x[t]);
}

// W [900,300] -> WT_bf [n][k] = W[k][n]
__global__ void conv_WT_kernel(const float* W, unsigned short* wt) {
  int t = blockIdx.x * blockDim.x + threadIdx.x;
  if (t >= 900 * Dd) return;
  int n = t / 900, k = t % 900;
  wt[(size_t)n * NEP + k] = f2bf(W[(size_t)k * Dd + n]);
}

// B [300,300] -> Bm_bf[d][e] = B[d][e]   (g = f @ B^T, rows=d, K=e)
__global__ void conv_B_kernel(const float* B, unsigned short* bm) {
  int t = blockIdx.x * blockDim.x + threadIdx.x;
  if (t >= Dd * Dd) return;
  int d = t / Dd, e = t % Dd;
  bm[(size_t)d * DP + e] = f2bf(B[t]);
}

// D [k][d][e] -> DT_bf[k][e][d]  (fD_k = f @ D_k : rows=e, K=d)
__global__ void conv_DT_kernel(const float* D, unsigned short* dt) {
  int t = blockIdx.x * blockDim.x + threadIdx.x;
  if (t >= Kk * Dd * Dd) return;
  int k = t / (Dd * Dd), r = t % (Dd * Dd);
  int e = r / Dd, d = r % Dd;
  dt[(size_t)k * COLP * DP + (size_t)e * DP + d] = f2bf(D[(size_t)(k * Dd + d) * Dd + e]);
}

// R [k][d][e] -> RT_bf[k][e][d]  (G_k = ent @ R_k : rows=e, K=d)
__global__ void conv_RT_kernel(const float* R, unsigned short* rt) {
  int t = blockIdx.x * blockDim.x + threadIdx.x;
  if (t >= Kk * Dd * Dd) return;
  int k = t / (Dd * Dd), r = t % (Dd * Dd);
  int e = r / Dd, d = r % Dd;
  rt[(size_t)k * COLP * DP + (size_t)e * DP + d] = f2bf(R[(size_t)(k * Dd + d) * Dd + e]);
}

// ---------------- GEMM kernels (8 waves/block, 1 tile/wave) ----------------
// f = tanh(fmc_in @ W + b) -> f_bf [ROWP][DP]
__global__ void gemm_f_kernel(const unsigned short* Abf, const unsigned short* Bbf,
                              const float* bias, unsigned short* fbf) {
  int wave = threadIdx.x >> 5, lane = threadIdx.x & 31;
  int tile = blockIdx.x * 8 + wave;
  if (tile >= T_ROW * T_COL) return;
  int tr = tile / T_COL, tc = tile % T_COL;
  v8f acc = wmma_nt_tile(Abf, NEP, Bbf, NEP, tr * 16, tc * 16, NEP, lane);
  int hi8 = (lane & 16) ? 8 : 0;
  int n = tc * 16 + (lane & 15);
  if (n < Dd) {
    float bn = bias[n];
#pragma unroll
    for (int v = 0; v < 8; ++v) {
      int m = tr * 16 + v + hi8;
      if (m < Mm) fbf[(size_t)m * DP + n] = f2bf(tanhf(acc[v] + bn));
    }
  }
}

// g = f @ B^T  (f32 out, [100][300])
__global__ void gemm_g_kernel(const unsigned short* fbf, const unsigned short* bm,
                              float* g) {
  int wave = threadIdx.x >> 5, lane = threadIdx.x & 31;
  int tile = blockIdx.x * 8 + wave;
  if (tile >= T_ROW * T_COL) return;
  int tr = tile / T_COL, tc = tile % T_COL;
  v8f acc = wmma_nt_tile(fbf, DP, bm, DP, tr * 16, tc * 16, DP, lane);
  int hi8 = (lane & 16) ? 8 : 0;
  int n = tc * 16 + (lane & 15);
  if (n < Dd) {
#pragma unroll
    for (int v = 0; v < 8; ++v) {
      int m = tr * 16 + v + hi8;
      if (m < Mm) g[(size_t)m * Dd + n] = acc[v];
    }
  }
}

// fD_k = f @ D_k  (bf16 out, [k][ROWP][DP])
__global__ void gemm_fD_kernel(const unsigned short* fbf, const unsigned short* dt,
                               unsigned short* fd) {
  int wave = threadIdx.x >> 5, lane = threadIdx.x & 31;
  int k = blockIdx.y;
  int tile = blockIdx.x * 8 + wave;
  if (tile >= T_ROW * T_COL) return;
  int tr = tile / T_COL, tc = tile % T_COL;
  const unsigned short* Bk = dt + (size_t)k * COLP * DP;
  v8f acc = wmma_nt_tile(fbf, DP, Bk, DP, tr * 16, tc * 16, DP, lane);
  int hi8 = (lane & 16) ? 8 : 0;
  int n = tc * 16 + (lane & 15);
  if (n < Dd) {
    unsigned short* out = fd + (size_t)k * ROWP * DP;
#pragma unroll
    for (int v = 0; v < 8; ++v) {
      int m = tr * 16 + v + hi8;
      if (m < Mm) out[(size_t)m * DP + n] = f2bf(acc[v]);
    }
  }
}

// s_k = (fD_k @ f^T) / sqrt(300)  (f32, [k][100][100])
__global__ void gemm_s_kernel(const unsigned short* fd, const unsigned short* fbf,
                              float* s) {
  int wave = threadIdx.x >> 5, lane = threadIdx.x & 31;
  int k = blockIdx.y;
  int tile = blockIdx.x * 8 + wave;
  if (tile >= T_ROW * T_ROW) return;
  int tr = tile / T_ROW, tc = tile % T_ROW;
  const unsigned short* Ak = fd + (size_t)k * ROWP * DP;
  v8f acc = wmma_nt_tile(Ak, DP, fbf, DP, tr * 16, tc * 16, DP, lane);
  const float scale = 0.05773502691896258f;   // 1/sqrt(300)
  int hi8 = (lane & 16) ? 8 : 0;
  int n = tc * 16 + (lane & 15);
  if (n < Mm) {
    float* out = s + (size_t)k * Mm * Mm;
#pragma unroll
    for (int v = 0; v < 8; ++v) {
      int m = tr * 16 + v + hi8;
      if (m < Mm) out[(size_t)m * Mm + n] = acc[v] * scale;
    }
  }
}

// G_k = ent_flat @ R_k  (bf16 out, [k][MCP][DP])
__global__ void gemm_G_kernel(const unsigned short* entbf, const unsigned short* rt,
                              unsigned short* G) {
  int wave = threadIdx.x >> 5, lane = threadIdx.x & 31;
  int k = blockIdx.y;
  int tile = blockIdx.x * 8 + wave;
  if (tile >= T_MC * T_COL) return;
  int tr = tile / T_COL, tc = tile % T_COL;
  const unsigned short* Bk = rt + (size_t)k * COLP * DP;
  v8f acc = wmma_nt_tile(entbf, DP, Bk, DP, tr * 16, tc * 16, DP, lane);
  int hi8 = (lane & 16) ? 8 : 0;
  int n = tc * 16 + (lane & 15);
  if (n < Dd) {
    unsigned short* out = G + (size_t)k * MCP * DP;
#pragma unroll
    for (int v = 0; v < 8; ++v) {
      int m = tr * 16 + v + hi8;
      if (m < MC) out[(size_t)m * DP + n] = f2bf(acc[v]);
    }
  }
}

// phi[i,j,p,q] = sum_k a[i,j,k] * (G_k @ ent_flat^T)[(j,q),(i,p)]
// Heavy GEMM: 3008x3008x320 x3 accumulators; a-scaling fused into epilogue.
__global__ void gemm_phi_kernel(const unsigned short* G, const unsigned short* entbf,
                                const float* a, float* phi) {
  int wave = threadIdx.x >> 5, lane = threadIdx.x & 31;
  int tile = blockIdx.x * 8 + wave;
  if (tile >= T_MC * T_MC) return;
  int tr = tile / T_MC, tc = tile % T_MC;
  int rowBase = tr * 16, colBase = tc * 16;
  const unsigned short* G0 = G;
  const unsigned short* G1 = G + (size_t)MCP * DP;
  const unsigned short* G2 = G + (size_t)2 * MCP * DP;
  v8f acc0 = {0.f, 0.f, 0.f, 0.f, 0.f, 0.f, 0.f, 0.f};
  v8f acc1 = acc0, acc2 = acc0;
  for (int kc = 0; kc < DP; kc += 32) {
    if (kc + 32 < DP) {  // keep the streams warm (global_prefetch_b8)
      __builtin_prefetch(entbf + (size_t)(colBase + (lane & 15)) * DP + kc + 32, 0, 0);
      __builtin_prefetch(G0 + (size_t)(rowBase + (lane & 15)) * DP + kc + 32, 0, 0);
    }
    v16bf bfr = load_b_frag(entbf, DP, colBase, kc, lane);
    v16bf a0 = load_a_frag(G0, DP, rowBase, kc, lane);
    acc0 = wmma_acc(a0, bfr, acc0);
    v16bf a1 = load_a_frag(G1, DP, rowBase, kc, lane);
    acc1 = wmma_acc(a1, bfr, acc1);
    v16bf a2 = load_a_frag(G2, DP, rowBase, kc, lane);
    acc2 = wmma_acc(a2, bfr, acc2);
  }
  int hi8 = (lane & 16) ? 8 : 0;
  int n = colBase + (lane & 15);          // (i,p) index
  if (n < MC) {
    int i = n / Cc, p = n % Cc;
#pragma unroll
    for (int v = 0; v < 8; ++v) {
      int m = rowBase + v + hi8;          // (j,q) index
      if (m < MC) {
        int j = m / Cc, q = m % Cc;
        const float* av = a + (size_t)(i * Mm + j) * Kk;
        float val = av[0] * acc0[v] + av[1] * acc1[v] + av[2] * acc2[v];
        phi[((size_t)(i * Mm + j) * Cc + p) * Cc + q] = val;
      }
    }
  }
}

// ---------------- small fp32 kernels ----------------
__global__ void psi_kernel(const float* ent, const float* g, float* psi) {
  int t = blockIdx.x * blockDim.x + threadIdx.x;   // (m,c)
  if (t >= Mm * Cc) return;
  int m = t / Cc;
  const float* er = ent + (size_t)t * Dd;
  const float* gr = g + (size_t)m * Dd;
  float s = 0.f;
  for (int d = 0; d < Dd; ++d) s += er[d] * gr[d];
  psi[t] = s;
}

__global__ void a_softmax_kernel(const float* s, float* a) {
  int t = blockIdx.x * blockDim.x + threadIdx.x;   // t = i*100 + j
  if (t >= Mm * Mm) return;
  float s0 = s[t], s1 = s[Mm * Mm + t], s2 = s[2 * Mm * Mm + t];
  float mx = fmaxf(s0, fmaxf(s1, s2));
  float e0 = expf(s0 - mx), e1 = expf(s1 - mx), e2 = expf(s2 - mx);
  float inv = 1.f / (e0 + e1 + e2);
  a[(size_t)t * Kk + 0] = e0 * inv;
  a[(size_t)t * Kk + 1] = e1 * inv;
  a[(size_t)t * Kk + 2] = e2 * inv;
}

__global__ void stot_kernel(const float* mbar, float* stot) {
  int t = blockIdx.x * blockDim.x + threadIdx.x;   // (i,p)
  if (t >= Mm * Cc) return;
  float s = 0.f;
  for (int j = 0; j < Mm; ++j) s += mbar[(size_t)j * Mm * Cc + t];
  stot[t] = s;
}

// one damped LBP update: block = (i,j), lane = q
__global__ void lbp_step_kernel(const float* psi, const float* stot, const float* phi,
                                const float* mcur, float* mnext) {
  int b = blockIdx.x;              // i*100 + j
  int i = b / Mm, j = b % Mm;
  int lane = threadIdx.x;
  __shared__ float base[32];
  if (lane < Cc)
    base[lane] = psi[i * Cc + lane] + stot[i * Cc + lane] -
                 mcur[(size_t)(j * Mm + i) * Cc + lane];
  __syncthreads();
  float mv = (lane < Cc) ? 0.0f : -1e30f;    // maxValue init 0 (relu)
  if (lane < Cc) {
    const float* ph = phi + (size_t)b * Cc * Cc + lane;   // + p*Cc
    for (int p = 0; p < Cc; ++p) mv = fmaxf(mv, base[p] + ph[(size_t)p * Cc]);
  }
  // softmax over q (lanes 0..29) via wave32 butterflies
  float xmax = mv;
  for (int off = 16; off > 0; off >>= 1) xmax = fmaxf(xmax, __shfl_xor(xmax, off, 32));
  float e = (lane < Cc) ? expf(mv - xmax) : 0.f;
  float es = e;
  for (int off = 16; off > 0; off >>= 1) es += __shfl_xor(es, off, 32);
  if (lane < Cc) {
    float soft = e / es;
    float mo = mcur[(size_t)b * Cc + lane];
    mnext[(size_t)b * Cc + lane] = logf(0.5f * expf(mo) + 0.5f * soft);
  }
}

__global__ void final_kernel(const float* psi, const float* mbar, float* out) {
  int i = blockIdx.x, lane = threadIdx.x;
  float u = -1e30f;
  if (lane < Cc) {
    float s = 0.f;
    for (int j = 0; j < Mm; ++j) s += mbar[(size_t)(j * Mm + i) * Cc + lane];
    u = psi[i * Cc + lane] + s - mbar[(size_t)(i * Mm + i) * Cc + lane];
  }
  float xmax = u;
  for (int off = 16; off > 0; off >>= 1) xmax = fmaxf(xmax, __shfl_xor(xmax, off, 32));
  float e = (lane < Cc) ? expf(u - xmax) : 0.f;
  float es = e;
  for (int off = 16; off > 0; off >>= 1) es += __shfl_xor(es, off, 32);
  if (lane < Cc) out[i * Cc + lane] = e / es;
}

// ---------------- launcher ----------------
extern "C" void kernel_launch(void* const* d_in, const int* in_sizes, int n_in,
                              void* d_out, int out_size, void* d_ws, size_t ws_size,
                              hipStream_t stream) {
  (void)in_sizes; (void)n_in; (void)out_size; (void)ws_size;
  const float* ent   = (const float*)d_in[0];
  const float* fmcin = (const float*)d_in[1];
  const float* Wf    = (const float*)d_in[2];
  const float* bias  = (const float*)d_in[3];
  const float* Bmat  = (const float*)d_in[4];
  const float* Rmat  = (const float*)d_in[5];
  const float* Dmat  = (const float*)d_in[6];
  float* out = (float*)d_out;
  char* ws = (char*)d_ws;

  size_t off = 0;
  unsigned short* entbf = (unsigned short*)(ws + off); off += 2ull * MCP * DP;
  unsigned short* Gbf   = (unsigned short*)(ws + off); off += 2ull * Kk * MCP * DP;
  unsigned short* fmcbf = (unsigned short*)(ws + off); off += 2ull * ROWP * NEP;
  unsigned short* wtbf  = (unsigned short*)(ws + off); off += 2ull * COLP * NEP;
  unsigned short* fbf   = (unsigned short*)(ws + off); off += 2ull * ROWP * DP;
  unsigned short* bmbf  = (unsigned short*)(ws + off); off += 2ull * COLP * DP;
  unsigned short* dtbf  = (unsigned short*)(ws + off); off += 2ull * Kk * COLP * DP;
  unsigned short* rtbf  = (unsigned short*)(ws + off); off += 2ull * Kk * COLP * DP;
  unsigned short* fdbf  = (unsigned short*)(ws + off); off += 2ull * Kk * ROWP * DP;
  size_t arenaBytes = off;
  float* gbuf = (float*)(ws + off); off += 4ull * Mm * Dd;
  float* psi  = (float*)(ws + off); off += 4ull * Mm * Cc;
  float* sbuf = (float*)(ws + off); off += 4ull * Kk * Mm * Mm;
  float* abuf = (float*)(ws + off); off += 4ull * Mm * Mm * Kk;
  float* phi  = (float*)(ws + off); off += 4ull * Mm * Mm * Cc * Cc;
  float* mb0  = (float*)(ws + off); off += 4ull * Mm * Mm * Cc;
  float* mb1  = (float*)(ws + off); off += 4ull * Mm * Mm * Cc;
  float* stot = (float*)(ws + off); off += 4ull * Mm * Cc;

  const int TPB = 256;
  // zero bf16 arena (gives zero-padding for all WMMA operand pads) + mbar0
  {
    int n = (int)(arenaBytes / 4);
    zero_u32_kernel<<<(n + TPB - 1) / TPB, TPB, 0, stream>>>((unsigned int*)ws, n);
    int n2 = Mm * Mm * Cc;
    zero_u32_kernel<<<(n2 + TPB - 1) / TPB, TPB, 0, stream>>>((unsigned int*)mb0, n2);
  }
  // fp32 -> bf16 conversions (with the needed transposes)
  conv_ent_kernel<<<(MC * Dd + TPB - 1) / TPB, TPB, 0, stream>>>(ent, entbf);
  conv_fmc_kernel<<<(Mm * 900 + TPB - 1) / TPB, TPB, 0, stream>>>(fmcin, fmcbf);
  conv_WT_kernel<<<(900 * Dd + TPB - 1) / TPB, TPB, 0, stream>>>(Wf, wtbf);
  conv_B_kernel<<<(Dd * Dd + TPB - 1) / TPB, TPB, 0, stream>>>(Bmat, bmbf);
  conv_DT_kernel<<<(Kk * Dd * Dd + TPB - 1) / TPB, TPB, 0, stream>>>(Dmat, dtbf);
  conv_RT_kernel<<<(Kk * Dd * Dd + TPB - 1) / TPB, TPB, 0, stream>>>(Rmat, rtbf);

  // f = tanh(fmc_in @ W + b)
  gemm_f_kernel<<<(T_ROW * T_COL + 7) / 8, TPB, 0, stream>>>(fmcbf, wtbf, bias, fbf);
  // g = f @ B^T ; psi = rowdot(ent, g)
  gemm_g_kernel<<<(T_ROW * T_COL + 7) / 8, TPB, 0, stream>>>(fbf, bmbf, gbuf);
  psi_kernel<<<(Mm * Cc + TPB - 1) / TPB, TPB, 0, stream>>>(ent, gbuf, psi);
  // s_k = (f @ D_k) @ f^T / sqrt(d) ; a = softmax_k
  gemm_fD_kernel<<<dim3((T_ROW * T_COL + 7) / 8, Kk), TPB, 0, stream>>>(fbf, dtbf, fdbf);
  gemm_s_kernel<<<dim3((T_ROW * T_ROW + 7) / 8, Kk), TPB, 0, stream>>>(fdbf, fbf, sbuf);
  a_softmax_kernel<<<(Mm * Mm + TPB - 1) / TPB, TPB, 0, stream>>>(sbuf, abuf);
  // G_k = ent @ R_k ; phi = sum_k a_ijk * (G_k @ ent^T)
  gemm_G_kernel<<<dim3((T_MC * T_COL + 7) / 8, Kk), TPB, 0, stream>>>(entbf, rtbf, Gbf);
  gemm_phi_kernel<<<(T_MC * T_MC + 7) / 8, TPB, 0, stream>>>(Gbf, entbf, abuf, phi);

  // 10 damped LBP iterations (ping-pong mbar buffers)
  float* cur = mb0;
  float* nxt = mb1;
  for (int it = 0; it < 10; ++it) {
    stot_kernel<<<(Mm * Cc + TPB - 1) / TPB, TPB, 0, stream>>>(cur, stot);
    lbp_step_kernel<<<Mm * Mm, 32, 0, stream>>>(psi, stot, phi, cur, nxt);
    float* t = cur; cur = nxt; nxt = t;
  }
  final_kernel<<<Mm, 32, 0, stream>>>(psi, cur, out);
}
